// ManifoldMHA_31327491457694
// MI455X (gfx1250) — matrix operand order repaired and checked by reading, not compiled, not run on hardware
//
#include <hip/hip_runtime.h>
#include <cmath>

#define HID 512
#define NH 8
#define SEQ 256
#define BATCH 2
#define ROWS (BATCH * SEQ)      /* 512 */
#define BH (BATCH * NH)         /* 16  */
#define CHUNKS (BH * SEQ)       /* 4096 */
#define EPS 1e-15f
#define MAXNORM 0.996f
#define ART_CLAMP 0.9999999f

typedef __attribute__((ext_vector_type(16))) __bf16 v16bf;
typedef __attribute__((ext_vector_type(8))) float v8f;
typedef __attribute__((ext_vector_type(4))) unsigned int v4u;
typedef __attribute__((ext_vector_type(8))) int v8i;
typedef __attribute__((ext_vector_type(4))) int v4i;

union V16U { v16bf v; uint4 q[2]; };

__device__ __forceinline__ unsigned short f2bf(float f) {
  unsigned int u = __float_as_uint(f);
  u += 0x7FFFu + ((u >> 16) & 1u);
  return (unsigned short)(u >> 16);
}
__device__ __forceinline__ float bf2f(unsigned short h) {
  return __uint_as_float(((unsigned int)h) << 16);
}
__device__ __forceinline__ float artanh_c(float x) {
  x = fminf(fmaxf(x, -ART_CLAMP), ART_CLAMP);
  return 0.5f * logf((1.0f + x) / (1.0f - x));
}

// ---------------------------------------------------------------------------
// fp32 -> bf16 hi/lo split (elementwise)
// ---------------------------------------------------------------------------
__global__ void split_fp32(const float* __restrict__ src,
                           unsigned short* __restrict__ hi,
                           unsigned short* __restrict__ lo, int n) {
  int i = blockIdx.x * blockDim.x + threadIdx.x;
  if (i >= n) return;
  float f = src[i];
  unsigned short h = f2bf(f);
  hi[i] = h;
  lo[i] = f2bf(f - bf2f(h));
}

// query is (S,B,HID); GEMM wants row m = b*SEQ+s -> split with transpose
__global__ void split_query(const float* __restrict__ query,
                            unsigned short* __restrict__ hi,
                            unsigned short* __restrict__ lo) {
  int i = blockIdx.x * blockDim.x + threadIdx.x;  // m*512 + c
  int m = i >> 9, c = i & 511;
  int b = m >> 8, s = m & 255;
  float f = query[(size_t)(s * BATCH + b) * HID + c];
  unsigned short h = f2bf(f);
  hi[i] = h;
  lo[i] = f2bf(f - bf2f(h));
}

// ---------------------------------------------------------------------------
// TDM: DMA a rows x cols bf16 2-D tile (row stride = stride elems) into LDS.
// D# per ISA 8.3/8.4: group0 = {count=1, lds_addr, global_addr, type=2},
// group1 = {data_size=2B, tensor dims, tile dims, dim0 stride}. 2-D tensor ->
// descriptor groups 2/3 zero. Block-uniform args -> SGPR descriptor.
// ---------------------------------------------------------------------------
#if __has_builtin(__builtin_amdgcn_tensor_load_to_lds)
#define HAVE_TDM 1
__device__ __forceinline__ void tdm_load_2d_bf16(unsigned lds_byte_off,
                                                 const unsigned short* gptr,
                                                 int rows, int cols, int stride) {
  unsigned long long ga = (unsigned long long)gptr;
  v4u g0;
  g0.x = 1u;                                            // count=1, user mode
  g0.y = lds_byte_off;                                  // lds_addr (bytes)
  g0.z = (unsigned)(ga & 0xFFFFFFFFu);                  // global_addr[31:0]
  g0.w = (unsigned)((ga >> 32) & 0x01FFFFFFu) | (2u << 30);  // addr[56:32]|type=2
  v8i g1;
  g1[0] = (int)(1u << 16);                              // data_size=1 (2 bytes)
  g1[1] = (int)((unsigned)cols << 16);                  // tensor_dim0[15:0]
  g1[2] = (int)(((unsigned)cols >> 16) | ((unsigned)rows << 16)); // d0 hi|d1 lo
  g1[3] = (int)((((unsigned)rows >> 16) & 0xFFFFu) | ((unsigned)cols << 16)); // d1 hi|tile_dim0
  g1[4] = (int)((unsigned)rows & 0xFFFFu);              // tile_dim1 (tile_dim2=0)
  g1[5] = stride;                                       // tensor_dim0_stride[31:0]
  g1[6] = 0;
  g1[7] = 0;
  v4i gz = {0, 0, 0, 0};
#if __clang_major__ >= 23
  v8i gz8 = {0, 0, 0, 0, 0, 0, 0, 0};
  __builtin_amdgcn_tensor_load_to_lds(g0, g1, gz, gz, gz8, 0);
#else
  __builtin_amdgcn_tensor_load_to_lds(g0, g1, gz, gz, 0);
#endif
}
#endif

// ---------------------------------------------------------------------------
// Generic batched bf16x3 GEMM:  C[m,n] = sum_k A[m,k] * B[n,k]   (row-major)
// One 16x16 tile per wave; ISA 7.12.2 layouts for 16-bit A(16x32)/B(32x16).
// ---------------------------------------------------------------------------
__global__ void gemm_bf16x3(const unsigned short* __restrict__ Ahi,
                            const unsigned short* __restrict__ Alo,
                            const unsigned short* __restrict__ Bhi,
                            const unsigned short* __restrict__ Blo,
                            float* __restrict__ C,
                            int M, int N, int K, int lda, int ldb, int ldc,
                            long long sA, long long sB, long long sC) {
  const long long batch = blockIdx.y;
  Ahi += batch * sA; Alo += batch * sA;
  Bhi += batch * sB; Blo += batch * sB;
  C   += batch * sC;
  const int lane = threadIdx.x & 31;
  const int wave = threadIdx.x >> 5;
  const int tn = N >> 4, tm = M >> 4;
  const int tile = blockIdx.x * (blockDim.x >> 5) + wave;
  if (tile >= tm * tn) return;
  const int m0 = (tile / tn) << 4, n0 = (tile % tn) << 4;
  const int lr = lane & 15, lh = lane >> 4;
  const unsigned short* arh = Ahi + (size_t)(m0 + lr) * lda;
  const unsigned short* arl = Alo + (size_t)(m0 + lr) * lda;
  const unsigned short* brh = Bhi + (size_t)(n0 + lr) * ldb;
  const unsigned short* brl = Blo + (size_t)(n0 + lr) * ldb;
  v8f acc = {};
  for (int kb = 0; kb < K; kb += 32) {
    V16U Ah, Al, Bh, Bl;
    const int ka = kb + lh * 8;   // A: lanes0-15 K=kb..7 & kb+16..23; lanes16-31 +8
    Ah.q[0] = *(const uint4*)(arh + ka);
    Ah.q[1] = *(const uint4*)(arh + ka + 16);
    Al.q[0] = *(const uint4*)(arl + ka);
    Al.q[1] = *(const uint4*)(arl + ka + 16);
    const int kq = kb + lh * 8;   // B: lanes0-15 K=kb..7 & kb+16..23; lanes16-31 +8
    Bh.q[0] = *(const uint4*)(brh + kq);
    Bh.q[1] = *(const uint4*)(brh + kq + 16);
    Bl.q[0] = *(const uint4*)(brl + kq);
    Bl.q[1] = *(const uint4*)(brl + kq + 16);
    acc = __builtin_amdgcn_wmma_f32_16x16x32_bf16(false, Ah.v, false, Bh.v,
                                                  (short)0, acc, false, false);
    acc = __builtin_amdgcn_wmma_f32_16x16x32_bf16(false, Ah.v, false, Bl.v,
                                                  (short)0, acc, false, false);
    acc = __builtin_amdgcn_wmma_f32_16x16x32_bf16(false, Al.v, false, Bh.v,
                                                  (short)0, acc, false, false);
  }
  float* cp = C + (size_t)(m0 + lh * 8) * ldc + (n0 + lr);
#pragma unroll
  for (int r = 0; r < 8; ++r) cp[(size_t)r * ldc] = acc[r];
}

// ---------------------------------------------------------------------------
// Projection GEMM (M=N=K=512): block = 8 waves sharing one 16-row A panel.
// A panel (16x512 hi + lo, 32 KB) staged into LDS once via the Tensor Data
// Mover; waves read WMMA A-fragments from LDS (ds_load_b128), B streams from
// global. TENSORcnt wait + workgroup barrier order the DMA against reads.
// ---------------------------------------------------------------------------
#define LDS_A_HALF (16 * 512)

__global__ void gemm_proj_tdm(const unsigned short* __restrict__ Ahi,
                              const unsigned short* __restrict__ Alo,
                              const unsigned short* __restrict__ Bhi,
                              const unsigned short* __restrict__ Blo,
                              float* __restrict__ C) {
  __shared__ unsigned short ldsA[2 * LDS_A_HALF];   // [hi | lo], 32 KB
  const int lane = threadIdx.x & 31;
  const int wave = threadIdx.x >> 5;
  const int mt = blockIdx.x >> 2;                   // 32 m-tiles
  const int ng = blockIdx.x & 3;                    // 4 n-groups x 8 waves
  const int m0 = mt << 4;
  const int n0 = (ng << 7) + (wave << 4);
#ifdef HAVE_TDM
  if (wave == 0) {
    tdm_load_2d_bf16(0u, Ahi + (size_t)m0 * 512, 16, 512, 512);
    tdm_load_2d_bf16((unsigned)(LDS_A_HALF * 2), Alo + (size_t)m0 * 512, 16, 512, 512);
    __builtin_amdgcn_s_wait_tensorcnt(0);
  }
#else
  for (int i = threadIdx.x; i < LDS_A_HALF; i += 256) {
    ldsA[i] = Ahi[(size_t)m0 * 512 + i];
    ldsA[LDS_A_HALF + i] = Alo[(size_t)m0 * 512 + i];
  }
#endif
  __syncthreads();
  const int lr = lane & 15, lh = lane >> 4;
  const unsigned short* lah = ldsA + (size_t)lr * 512;
  const unsigned short* lal = ldsA + LDS_A_HALF + (size_t)lr * 512;
  const unsigned short* brh = Bhi + (size_t)(n0 + lr) * 512;
  const unsigned short* brl = Blo + (size_t)(n0 + lr) * 512;
  v8f acc = {};
  for (int kb = 0; kb < 512; kb += 32) {
    V16U Ah, Al, Bh, Bl;
    const int ka = kb + lh * 8;
    Ah.q[0] = *(const uint4*)(lah + ka);
    Ah.q[1] = *(const uint4*)(lah + ka + 16);
    Al.q[0] = *(const uint4*)(lal + ka);
    Al.q[1] = *(const uint4*)(lal + ka + 16);
    const int kq = kb + lh * 8;
    Bh.q[0] = *(const uint4*)(brh + kq);
    Bh.q[1] = *(const uint4*)(brh + kq + 16);
    Bl.q[0] = *(const uint4*)(brl + kq);
    Bl.q[1] = *(const uint4*)(brl + kq + 16);
    acc = __builtin_amdgcn_wmma_f32_16x16x32_bf16(false, Ah.v, false, Bh.v,
                                                  (short)0, acc, false, false);
    acc = __builtin_amdgcn_wmma_f32_16x16x32_bf16(false, Ah.v, false, Bl.v,
                                                  (short)0, acc, false, false);
    acc = __builtin_amdgcn_wmma_f32_16x16x32_bf16(false, Al.v, false, Bh.v,
                                                  (short)0, acc, false, false);
  }
  float* cp = C + (size_t)(m0 + lh * 8) * 512 + (n0 + lr);
#pragma unroll
  for (int r = 0; r < 8; ++r) cp[(size_t)r * 512] = acc[r];
}

// ---------------------------------------------------------------------------
// mobius_matvec tail + mobius_add(bias) + projx + logmap0*scale, per 512-row
// ---------------------------------------------------------------------------
__global__ void rowpost(const float* __restrict__ query,
                        const float* __restrict__ mxq, const float* __restrict__ mxk,
                        const float* __restrict__ mxv,
                        const float* __restrict__ bq, const float* __restrict__ bk,
                        const float* __restrict__ bv,
                        float* __restrict__ oq, float* __restrict__ ok,
                        float* __restrict__ ov, float scale) {
  __shared__ float sm[4][256];
  const int m = blockIdx.x, p = blockIdx.y, t = threadIdx.x;
  const float* mx   = (p == 0) ? mxq : (p == 1) ? mxk : mxv;
  const float* bias = (p == 0) ? bq  : (p == 1) ? bk  : bv;
  float* out        = (p == 0) ? oq  : (p == 1) ? ok  : ov;
  const int b = m >> 8, s = m & 255;
  const float* xr = query + (size_t)(s * BATCH + b) * HID;
  const float* mr = mx + (size_t)m * HID;
  float x0 = xr[t], x1 = xr[t + 256];
  float v0 = mr[t], v1 = mr[t + 256];
  float b0 = bias[t], b1 = bias[t + 256];
  sm[0][t] = x0 * x0 + x1 * x1;
  sm[1][t] = v0 * v0 + v1 * v1;
  sm[2][t] = v0 * b0 + v1 * b1;
  sm[3][t] = b0 * b0 + b1 * b1;
  __syncthreads();
  for (int off = 128; off > 0; off >>= 1) {
    if (t < off) {
      sm[0][t] += sm[0][t + off]; sm[1][t] += sm[1][t + off];
      sm[2][t] += sm[2][t + off]; sm[3][t] += sm[3][t + off];
    }
    __syncthreads();
  }
  float x2 = sm[0][0], mx2 = sm[1][0], mxb = sm[2][0], b2 = sm[3][0];
  __syncthreads();
  float xn = fmaxf(sqrtf(x2), EPS);
  float mxn_raw = sqrtf(mx2);
  float mxn = fmaxf(mxn_raw, EPS);
  float tt = tanhf(mxn / xn * artanh_c(xn));   // ||res|| = |tt|
  if (mxn_raw <= 0.0f) tt = 0.0f;              // all(mx==0) -> res = 0
  float rf = tt / mxn;                          // res_i = rf * mx_i
  float res2 = tt * tt;
  float rb = rf * mxb;                          // res . bias
  float c1 = 1.0f + 2.0f * rb + b2;             // K = -1
  float c2 = 1.0f - res2;
  float den = fmaxf(1.0f + 2.0f * rb + res2 * b2, EPS);
  float y0 = (c1 * rf * v0 + c2 * b0) / den;
  float y1 = (c1 * rf * v1 + c2 * b1) / den;
  sm[0][t] = y0 * y0 + y1 * y1;
  __syncthreads();
  for (int off = 128; off > 0; off >>= 1) {
    if (t < off) sm[0][t] += sm[0][t + off];
    __syncthreads();
  }
  float yn = sqrtf(sm[0][0]);
  float pf = (yn > MAXNORM) ? (MAXNORM / yn) : 1.0f;   // projx
  y0 *= pf; y1 *= pf;
  float n = fmaxf(fminf(yn, MAXNORM), EPS);
  float lf = artanh_c(n) / n * scale;                  // logmap0 * scale
  float* orow = out + (size_t)m * HID;
  orow[t] = y0 * lf;
  orow[t + 256] = y1 * lf;
}

// ---------------------------------------------------------------------------
// expmap0 per 64-chunk (chunk order == (b*8+h)*256+s); emit bf16 hi/lo + sq,
// v-branch also emits gamma and transposed gamma*v (for the midpoint GEMM).
// ---------------------------------------------------------------------------
__global__ void expmap64(const float* __restrict__ ql, const float* __restrict__ kl,
                         const float* __restrict__ vl,
                         unsigned short* __restrict__ qhi, unsigned short* __restrict__ qlo,
                         float* __restrict__ sqq,
                         unsigned short* __restrict__ khi, unsigned short* __restrict__ klo,
                         float* __restrict__ sqk,
                         float* __restrict__ gamma,
                         unsigned short* __restrict__ gvth, unsigned short* __restrict__ gvtl) {
  __shared__ float sm[64];
  const int g = blockIdx.x, p = blockIdx.y, d = threadIdx.x;
  const float* src = (p == 0) ? ql : (p == 1) ? kl : vl;
  float u = src[(size_t)g * 64 + d];
  sm[d] = u * u;
  __syncthreads();
  for (int off = 32; off > 0; off >>= 1) {
    if (d < off) sm[d] += sm[d + off];
    __syncthreads();
  }
  float n = fmaxf(sqrtf(sm[0]), EPS);
  float tn = tanhf(n);
  float w = tn / n * u;                               // expmap0
  float pf = (tn > MAXNORM) ? (MAXNORM / tn) : 1.0f;  // projx
  w *= pf;
  float postn = fminf(tn, MAXNORM);
  float sq = postn * postn;
  size_t gi = (size_t)g * 64 + d;
  if (p == 0) {
    unsigned short h = f2bf(w);
    qhi[gi] = h; qlo[gi] = f2bf(w - bf2f(h));
    if (d == 0) sqq[g] = sq;
  } else if (p == 1) {
    unsigned short h = f2bf(w);
    khi[gi] = h; klo[gi] = f2bf(w - bf2f(h));
    if (d == 0) sqk[g] = sq;
  } else {
    float gam = 2.0f / fmaxf(1.0f - sq, EPS);   // lambda_x, K = -1
    if (d == 0) gamma[g] = gam;
    float gv = gam * w;
    unsigned short h = f2bf(gv);
    int bh = g >> 8, s = g & 255;
    size_t ti = ((size_t)bh * 64 + d) * 256 + s;  // transpose: (bh, d, k)
    gvth[ti] = h; gvtl[ti] = f2bf(gv - bf2f(h));
  }
}

// scores from Gram + norms: closed form of ||mobius_add(-q,k)||
__global__ void scores_kernel(float* __restrict__ sc, const float* __restrict__ sqq,
                              const float* __restrict__ sqk) {
  int i = blockIdx.x * blockDim.x + threadIdx.x;   // (bh, q, k)
  int bh = i >> 16, qi = (i >> 8) & 255, ki = i & 255;
  float g = sc[i];                                  // q . k
  float q2 = sqq[bh * 256 + qi], k2 = sqk[bh * 256 + ki];
  float c1 = 1.0f - 2.0f * g + k2;                  // 1 + 2*(x.y) + y2, x = -q
  float c2 = 1.0f - q2;
  float num2 = c1 * c1 * q2 - 2.0f * c1 * c2 * g + c2 * c2 * k2;
  float den = fmaxf(1.0f - 2.0f * g + q2 * k2, EPS);
  float t1 = sqrtf(fmaxf(num2, 0.0f)) / den;
  float s = -2.0f * artanh_c(t1);
  sc[i] = fminf(fmaxf(s, -1e10f), 1e10f);
}

// softmax over the QUERY axis (dim=-2), per (bh, k); coalesced across k
__global__ void softmax_q(const float* __restrict__ sc, float* __restrict__ probs) {
  int bh = blockIdx.x, k = threadIdx.x;
  const float* base = sc + (size_t)bh * 65536 + k;
  float mx = -3.402823e38f;
  for (int q = 0; q < 256; ++q) mx = fmaxf(mx, base[q * 256]);
  float sum = 0.0f;
  for (int q = 0; q < 256; ++q) sum += expf(base[q * 256] - mx);
  float inv = 1.0f / sum;
  float* ob = probs + (size_t)bh * 65536 + k;
  for (int q = 0; q < 256; ++q) ob[q * 256] = expf(base[q * 256] - mx) * inv;
}

__global__ void denom_kernel(const float* __restrict__ probs, const float* __restrict__ gamma,
                             float* __restrict__ den) {
  __shared__ float sm[256];
  int gq = blockIdx.x;        // bh*256 + q
  int bh = gq >> 8, k = threadIdx.x;
  sm[k] = probs[(size_t)gq * 256 + k] * (gamma[bh * 256 + k] - 1.0f);
  __syncthreads();
  for (int off = 128; off > 0; off >>= 1) {
    if (k < off) sm[k] += sm[k + off];
    __syncthreads();
  }
  if (k == 0) den[gq] = sm[0];
}

// two_mean -> ctx -> logmap0 * invscale, per 64-chunk
__global__ void ctx64(const float* __restrict__ nom, const float* __restrict__ den,
                      float* __restrict__ lctx, float invscale) {
  __shared__ float sm[64];
  int g = blockIdx.x, d = threadIdx.x;
  float dn = fmaxf(den[g], 1e-10f);
  float tm = nom[(size_t)g * 64 + d] / dn;
  sm[d] = tm * tm;
  __syncthreads();
  for (int off = 32; off > 0; off >>= 1) {
    if (d < off) sm[d] += sm[d + off];
    __syncthreads();
  }
  float sq = sm[0];
  float fac = 1.0f / (1.0f + sqrtf(fmaxf(1.0f - sq, 1e-10f)));
  float c = tm * fac;
  float cn = fmaxf(fac * sqrtf(sq), EPS);
  lctx[(size_t)g * 64 + d] = c * (artanh_c(cn) / cn) * invscale;
}

// gather heads -> 512-dim expmap0 + projx -> (S,B,512) output
__global__ void final_kernel(const float* __restrict__ lctx, float* __restrict__ out) {
  __shared__ float sm[256];
  int m = blockIdx.x, t = threadIdx.x;
  int b = m >> 8, s = m & 255;
  int h0 = t >> 6, d0 = t & 63;
  int c1i = t + 256, h1 = c1i >> 6, d1 = c1i & 63;
  float u0 = lctx[((size_t)(b * NH + h0) * SEQ + s) * 64 + d0];
  float u1 = lctx[((size_t)(b * NH + h1) * SEQ + s) * 64 + d1];
  sm[t] = u0 * u0 + u1 * u1;
  __syncthreads();
  for (int off = 128; off > 0; off >>= 1) {
    if (t < off) sm[t] += sm[t + off];
    __syncthreads();
  }
  float n = fmaxf(sqrtf(sm[0]), EPS);
  float tn = tanhf(n);
  float f = tn / n;
  if (tn > MAXNORM) f *= MAXNORM / tn;
  float* orow = out + (size_t)(s * BATCH + b) * HID;
  orow[t] = u0 * f;
  orow[t + 256] = u1 * f;
}

extern "C" void kernel_launch(void* const* d_in, const int* in_sizes, int n_in,
                              void* d_out, int out_size, void* d_ws, size_t ws_size,
                              hipStream_t stream) {
  const float* query = (const float*)d_in[0];
  const float* Wq = (const float*)d_in[1];
  const float* bq = (const float*)d_in[2];
  const float* Wk = (const float*)d_in[3];
  const float* bk = (const float*)d_in[4];
  const float* Wv = (const float*)d_in[5];
  const float* bv = (const float*)d_in[6];
  float* out_ctx = (float*)d_out;                              // (S,B,512)
  float* out_probs = out_ctx + (size_t)SEQ * BATCH * HID;      // (B,H,S,S)

  char* ws = (char*)d_ws;
  size_t off = 0;
  auto alloc = [&](size_t bytes) {
    size_t o = off;
    off += (bytes + 255) & ~(size_t)255;
    return (void*)(ws + o);
  };
  const size_t NBF = (size_t)ROWS * HID;        // 262144
  unsigned short* xhi = (unsigned short*)alloc(NBF * 2);
  unsigned short* xlo = (unsigned short*)alloc(NBF * 2);
  unsigned short* wh[3], *wl[3];
  float* mx[3];
  float* lv[3];
  for (int p = 0; p < 3; ++p) {
    wh[p] = (unsigned short*)alloc(NBF * 2);
    wl[p] = (unsigned short*)alloc(NBF * 2);
    mx[p] = (float*)alloc(NBF * 4);
    lv[p] = (float*)alloc(NBF * 4);
  }
  const size_t NHD = (size_t)CHUNKS * 64;       // 262144
  unsigned short* qhi = (unsigned short*)alloc(NHD * 2);
  unsigned short* qlo = (unsigned short*)alloc(NHD * 2);
  unsigned short* khi = (unsigned short*)alloc(NHD * 2);
  unsigned short* klo = (unsigned short*)alloc(NHD * 2);
  float* sqq = (float*)alloc(CHUNKS * 4);
  float* sqk = (float*)alloc(CHUNKS * 4);
  float* gamma = (float*)alloc(CHUNKS * 4);
  unsigned short* gvth = (unsigned short*)alloc(NHD * 2);
  unsigned short* gvtl = (unsigned short*)alloc(NHD * 2);
  float* scores = (float*)alloc((size_t)BH * 65536 * 4);
  unsigned short* phi = (unsigned short*)alloc((size_t)BH * 65536 * 2);
  unsigned short* plo = (unsigned short*)alloc((size_t)BH * 65536 * 2);
  float* nom = (float*)alloc(NHD * 4);
  float* den = (float*)alloc(CHUNKS * 4);
  float* lctx = (float*)alloc(NHD * 4);

  auto betaf = [](double a, double b) {
    return std::exp(std::lgamma(a) + std::lgamma(b) - std::lgamma(a + b));
  };
  double beta_ni = betaf(HID / (2.0 * NH), 0.5);   // beta(32, 0.5)
  double beta_n  = betaf(HID / 2.0, 0.5);          // beta(256, 0.5)
  float scale = (float)(beta_ni / beta_n);
  float invscale = (float)(beta_n / beta_ni);

  const float* Ws[3] = {Wq, Wk, Wv};
  const int n262k = (int)NBF;

  // 1) bf16 hi/lo splits of x (with (S,B)->(B,S) transpose) and weights
  split_query<<<1024, 256, 0, stream>>>(query, xhi, xlo);
  for (int p = 0; p < 3; ++p)
    split_fp32<<<1024, 256, 0, stream>>>(Ws[p], wh[p], wl[p], n262k);

  // 2) projections: mx = x @ W^T (M=N=K=512); TDM-staged A panel + bf16x3 WMMA
  for (int p = 0; p < 3; ++p)
    gemm_proj_tdm<<<128, 256, 0, stream>>>(xhi, xlo, wh[p], wl[p], mx[p]);

  // 3) mobius tail + bias + projx + logmap0*scale per row
  rowpost<<<dim3(ROWS, 3), 256, 0, stream>>>(query, mx[0], mx[1], mx[2],
                                             bq, bk, bv, lv[0], lv[1], lv[2], scale);

  // 4) expmap0 per head-chunk; bf16 splits, norms, gamma, gamma*v transpose
  expmap64<<<dim3(CHUNKS, 3), 64, 0, stream>>>(lv[0], lv[1], lv[2],
                                               qhi, qlo, sqq, khi, klo, sqk,
                                               gamma, gvth, gvtl);

  // 5) batched Gram: Q @ K^T per (b,h)  (M=256,N=256,K=64)
  gemm_bf16x3<<<dim3(32, BH), 256, 0, stream>>>(qhi, qlo, khi, klo, scores,
                                                256, 256, 64, 64, 64, 256,
                                                16384, 16384, 65536);

  // 6) closed-form mobius distance scores, softmax over q axis -> probs output
  scores_kernel<<<4096, 256, 0, stream>>>(scores, sqq, sqk);
  softmax_q<<<BH, 256, 0, stream>>>(scores, out_probs);

  // 7) gyromidpoint: denom reduction + nom = probs @ (gamma*v)
  split_fp32<<<4096, 256, 0, stream>>>(out_probs, phi, plo, BH * 65536);
  denom_kernel<<<CHUNKS, 256, 0, stream>>>(out_probs, gamma, den);
  gemm_bf16x3<<<dim3(8, BH), 256, 0, stream>>>(phi, plo, gvth, gvtl, nom,
                                               256, 64, 256, 256, 256, 64,
                                               65536, 16384, 16384);

  // 8) context + logmap0*invscale, then 512-dim expmap0 + output transpose
  ctx64<<<CHUNKS, 64, 0, stream>>>(nom, den, lctx, invscale);
  final_kernel<<<ROWS, 256, 0, stream>>>(lctx, out_ctx);
}
